// Decoder_910533066914
// MI455X (gfx1250) — matrix operand order repaired and checked by reading, not compiled, not run on hardware
//
#include <hip/hip_runtime.h>
#include <hip/hip_bf16.h>
#include <math.h>

// ---------------- CDNA5 WMMA types ----------------
typedef __attribute__((ext_vector_type(16))) __bf16 v16bf_t;
typedef __attribute__((ext_vector_type(8)))  __bf16 v8bf_t;
typedef __attribute__((ext_vector_type(8)))  float  v8f_t;

#define DEVFN static __device__ __forceinline__

DEVFN unsigned short f2bf(float f) {      // round-to-nearest-even fp32 -> bf16
  unsigned int u = __float_as_uint(f);
  u += 0x7FFFu + ((u >> 16) & 1u);
  return (unsigned short)(u >> 16);
}
DEVFN float sigm(float x) { return 1.0f / (1.0f + __expf(-x)); }

static constexpr int B_ = 32, T_ = 64, S_ = 64, D_ = 1024, A_ = 512, E_ = 256,
                     L_ = 1024, V_ = 12000, FL_ = 4096, KX_ = 1280; // KX = L + E

// =====================================================================
// bf16 GEMM: D[M,N] = A[M,K](lda) * BT[N,K]^T + C0
// c0mode: 0=none, 1=row bias C0[N], 2=full matrix C0[M,N]
// One wave32 per 32x16 output strip (two 16x16 M-tiles): each B-fragment
// load feeds TWO v_wmma_f32_16x16x32_bf16, halving B traffic vs 1 tile/wave.
// M mult of 32, N mult of 16, K mult of 32.
// =====================================================================
__global__ __launch_bounds__(256) void k_gemm_bf16(
    const unsigned short* __restrict__ A, int lda,
    const unsigned short* __restrict__ BT,
    const float* __restrict__ C0, int c0mode,
    float* __restrict__ D, int M, int N, int K)
{
  const int gwave  = (int)((blockIdx.x * blockDim.x + threadIdx.x) >> 5);
  const int lane   = (int)(threadIdx.x & 31);
  const int tilesN = N >> 4;
  const int tilesT = (M >> 5) * tilesN;
  if (gwave >= tilesT) return;               // whole wave exits together (EXEC all-1 for WMMA)
  const int m0  = (gwave / tilesN) << 5;     // 32-row strip
  const int n0  = (gwave % tilesN) << 4;
  const int l15 = lane & 15;
  const int hi  = lane >> 4;

  // A-fragments: rows m0+l15 and m0+16+l15; lanes<16 hold K {0..7,16..23}, lanes>=16 {8..15,24..31}
  const unsigned short* pa0 = A + (size_t)(m0 + l15) * lda + (hi << 3);
  const unsigned short* pa1 = pa0 + (size_t)16 * lda;
  // B-fragment: col = n0+l15; lanes<16 hold K {0..7,16..23}, lanes>=16 {8..15,24..31}
  const unsigned short* pb  = BT + (size_t)(n0 + l15) * K + (hi << 3);

  const int col    = n0 + l15;
  const int rbase0 = m0 + (hi << 3);         // C/D: VGPR r -> row rbase + r
  const int rbase1 = rbase0 + 16;

  v8f_t acc0, acc1;
  if (c0mode == 2) {
#pragma unroll
    for (int r = 0; r < 8; ++r) {
      acc0[r] = C0[(size_t)(rbase0 + r) * N + col];
      acc1[r] = C0[(size_t)(rbase1 + r) * N + col];
    }
  } else if (c0mode == 1) {
    const float bv = C0[col];
#pragma unroll
    for (int r = 0; r < 8; ++r) { acc0[r] = bv; acc1[r] = bv; }
  } else {
#pragma unroll
    for (int r = 0; r < 8; ++r) { acc0[r] = 0.0f; acc1[r] = 0.0f; }
  }

#pragma unroll 2
  for (int k0 = 0; k0 < K; k0 += 32) {
    union { v16bf_t v; v8bf_t h[2]; } a0, a1;
    a0.h[0] = *reinterpret_cast<const v8bf_t*>(pa0 + k0);
    a0.h[1] = *reinterpret_cast<const v8bf_t*>(pa0 + k0 + 16);
    a1.h[0] = *reinterpret_cast<const v8bf_t*>(pa1 + k0);
    a1.h[1] = *reinterpret_cast<const v8bf_t*>(pa1 + k0 + 16);
    union { v16bf_t v; v8bf_t h[2]; } bu;
    bu.h[0] = *reinterpret_cast<const v8bf_t*>(pb + k0);
    bu.h[1] = *reinterpret_cast<const v8bf_t*>(pb + k0 + 16);
    const v16bf_t b = bu.v;
    __builtin_prefetch(pb + k0 + 128, 0, 1);   // pull next B chunk toward WGP (global_prefetch)
    acc0 = __builtin_amdgcn_wmma_f32_16x16x32_bf16(false, a0.v, false, b,
                                                   (short)0, acc0, false, false);
    acc1 = __builtin_amdgcn_wmma_f32_16x16x32_bf16(false, a1.v, false, b,
                                                   (short)0, acc1, false, false);
  }

#pragma unroll
  for (int r = 0; r < 8; ++r) {
    D[(size_t)(rbase0 + r) * N + col] = acc0[r];
    D[(size_t)(rbase1 + r) * N + col] = acc1[r];
  }
}

// ---- fp32 -> bf16, transposing: out[n*out_ld + k] = bf16(in[k*in_ld + n])
__global__ void k_convT(const float* __restrict__ in, unsigned short* __restrict__ out,
                        int K, int N, int in_ld, int out_ld)
{
  int idx = blockIdx.x * blockDim.x + threadIdx.x;
  if (idx >= N * K) return;
  int n = idx / K, k = idx - n * K;
  out[(size_t)n * out_ld + k] = f2bf(in[(size_t)k * in_ld + n]);
}

// ---- fp32 -> bf16, flat copy
__global__ void k_conv(const float* __restrict__ in, unsigned short* __restrict__ out, int n)
{
  int idx = blockIdx.x * blockDim.x + threadIdx.x;
  if (idx < n) out[idx] = f2bf(in[idx]);
}

// ---- g[m] = opt_proj[m,:] . Va   (m over B*S)
__global__ void k_dotVa(const float* __restrict__ proj, const float* __restrict__ Va,
                        float* __restrict__ g)
{
  int m = blockIdx.x * blockDim.x + threadIdx.x;
  if (m >= B_ * S_) return;
  const float* p = proj + (size_t)m * A_;
  float s = 0.f;
  for (int a = 0; a < A_; ++a) s += p[a] * Va[a];
  g[m] = s;
}

// ---- step-invariant attention weights: w[b,s] = softmax_s(g[b,s]); also final attention out
__global__ void k_attn_softmax(const float* __restrict__ g, float* __restrict__ w,
                               float* __restrict__ attn_out)
{
  __shared__ float sm[S_];
  int b = blockIdx.x, s = threadIdx.x;
  float v = g[b * S_ + s];
  sm[s] = v; __syncthreads();
  for (int off = S_ / 2; off > 0; off >>= 1) {
    if (s < off) sm[s] = fmaxf(sm[s], sm[s + off]);
    __syncthreads();
  }
  float mx = sm[0]; __syncthreads();
  float e = __expf(v - mx);
  sm[s] = e; __syncthreads();
  for (int off = S_ / 2; off > 0; off >>= 1) {
    if (s < off) sm[s] += sm[s + off];
    __syncthreads();
  }
  float ww = e / sm[0];
  w[b * S_ + s] = ww;
  attn_out[b * S_ + s] = ww;
}

// ---- context[b,d] = sum_s w[b,s] * options[b,s,d]  (bf16 A-row for the zc GEMM)
__global__ void k_context(const float* __restrict__ w, const float* __restrict__ options,
                          unsigned short* __restrict__ ctx_bf)
{
  int idx = blockIdx.x * blockDim.x + threadIdx.x;
  if (idx >= B_ * D_) return;
  int b = idx >> 10, d = idx & (D_ - 1);
  float s = 0.f;
  for (int si = 0; si < S_; ++si)
    s += w[b * S_ + si] * options[(size_t)(b * S_ + si) * D_ + d];
  ctx_bf[idx] = f2bf(s);
}

// ---- init recurrent state; xa is fused A-row [h(1024) | emb(256)] in bf16
__global__ void k_init(const float* __restrict__ h0, const float* __restrict__ c0,
                       unsigned short* __restrict__ xa, float* __restrict__ hbuf,
                       float* __restrict__ cbuf)
{
  int idx = blockIdx.x * blockDim.x + threadIdx.x;
  if (idx >= B_ * L_) return;
  int b = idx >> 10, j = idx & (L_ - 1);
  float h = h0[idx];
  xa[b * KX_ + j] = f2bf(h);
  hbuf[idx] = h;
  cbuf[idx] = c0[idx];
}

// ---- gather emb[targets[:,t]] into tail of xa
__global__ void k_gather_emb(const float* __restrict__ emb, const int* __restrict__ targets,
                             int t, unsigned short* __restrict__ xa)
{
  int idx = blockIdx.x * blockDim.x + threadIdx.x;
  if (idx >= B_ * E_) return;
  int b = idx >> 8, e = idx & (E_ - 1);
  int id = targets[b * T_ + t];
  xa[b * KX_ + L_ + e] = f2bf(emb[(size_t)id * E_ + e]);
}

// ---- LSTM cell (Keras gate order i,f,g,o); writes fp32 h/c and bf16 h into xa
__global__ void k_lstm(const float* __restrict__ z, float* __restrict__ cbuf,
                       float* __restrict__ hbuf, unsigned short* __restrict__ xa)
{
  int idx = blockIdx.x * blockDim.x + threadIdx.x;
  if (idx >= B_ * L_) return;
  int b = idx >> 10, j = idx & (L_ - 1);
  const float* zb = z + (size_t)b * FL_;
  float zi = zb[j], zf = zb[L_ + j], zg = zb[2 * L_ + j], zo = zb[3 * L_ + j];
  float c = sigm(zf) * cbuf[idx] + sigm(zi) * tanhf(zg);
  float h = sigm(zo) * tanhf(c);
  cbuf[idx] = c;
  hbuf[idx] = h;
  xa[b * KX_ + j] = f2bf(h);
}

// ---- softmax over V=12000, write probs into d_out[b, t, :]
__global__ __launch_bounds__(256) void k_vsoftmax(const float* __restrict__ logits,
                                                  float* __restrict__ out, int t)
{
  __shared__ float sm[256];
  int b = blockIdx.x, tid = threadIdx.x;
  const float* lb = logits + (size_t)b * V_;
  float mx = -3.4e38f;
  for (int v = tid; v < V_; v += 256) mx = fmaxf(mx, lb[v]);
  sm[tid] = mx; __syncthreads();
  for (int off = 128; off > 0; off >>= 1) { if (tid < off) sm[tid] = fmaxf(sm[tid], sm[tid + off]); __syncthreads(); }
  mx = sm[0]; __syncthreads();
  float sum = 0.f;
  for (int v = tid; v < V_; v += 256) sum += __expf(lb[v] - mx);
  sm[tid] = sum; __syncthreads();
  for (int off = 128; off > 0; off >>= 1) { if (tid < off) sm[tid] += sm[tid + off]; __syncthreads(); }
  float inv = 1.0f / sm[0];
  float* ob = out + ((size_t)b * T_ + t) * V_;
  for (int v = tid; v < V_; v += 256) ob[v] = __expf(lb[v] - mx) * inv;
}

// ---- copy final h, c into d_out
__global__ void k_final(const float* __restrict__ hbuf, const float* __restrict__ cbuf,
                        float* __restrict__ oh, float* __restrict__ oc)
{
  int idx = blockIdx.x * blockDim.x + threadIdx.x;
  if (idx >= B_ * L_) return;
  oh[idx] = hbuf[idx];
  oc[idx] = cbuf[idx];
}

// =====================================================================
extern "C" void kernel_launch(void* const* d_in, const int* in_sizes, int n_in,
                              void* d_out, int out_size, void* d_ws, size_t ws_size,
                              hipStream_t stream)
{
  (void)in_sizes; (void)n_in; (void)out_size; (void)ws_size;

  const int*   targets = (const int*)  d_in[0];
  const float* options = (const float*)d_in[1];
  const float* h0      = (const float*)d_in[2];
  const float* c0      = (const float*)d_in[3];
  const float* emb     = (const float*)d_in[4];
  // d_in[5]=W1, d_in[6]=b1, d_in[10]=bVa: mathematically dead (softmax shift-invariance)
  const float* W2      = (const float*)d_in[7];
  const float* b2      = (const float*)d_in[8];
  const float* Va      = (const float*)d_in[9];
  const float* Wx      = (const float*)d_in[11];
  const float* Wh      = (const float*)d_in[12];
  const float* bl      = (const float*)d_in[13];
  const float* Wd      = (const float*)d_in[14];
  const float* bd      = (const float*)d_in[15];

  char* ws = (char*)d_ws;
  size_t off = 0;
  auto carve = [&](size_t bytes) -> void* {
    void* p = ws + off;
    off += (bytes + 255) & ~(size_t)255;
    return p;
  };

  unsigned short* W2T    = (unsigned short*)carve((size_t)A_ * D_ * 2);   // [A,D]
  unsigned short* WcombT = (unsigned short*)carve((size_t)FL_ * KX_ * 2); // [4L, L+E] = [Wh^T | WxE^T]
  unsigned short* WxCT   = (unsigned short*)carve((size_t)FL_ * D_ * 2);  // [4L, D]
  unsigned short* WdT    = (unsigned short*)carve((size_t)V_ * L_ * 2);   // [V, L]
  unsigned short* Aopt   = (unsigned short*)carve((size_t)B_ * S_ * D_ * 2);
  float* proj   = (float*)carve((size_t)B_ * S_ * A_ * 4);
  float* gbuf   = (float*)carve((size_t)B_ * S_ * 4);
  float* wbuf   = (float*)carve((size_t)B_ * S_ * 4);
  unsigned short* ctxbf = (unsigned short*)carve((size_t)B_ * D_ * 2);
  float* zc     = (float*)carve((size_t)B_ * FL_ * 4);
  unsigned short* xa = (unsigned short*)carve((size_t)B_ * KX_ * 2);
  float* hbuf   = (float*)carve((size_t)B_ * L_ * 4);
  float* cbuf   = (float*)carve((size_t)B_ * L_ * 4);
  float* zbuf   = (float*)carve((size_t)B_ * FL_ * 4);
  float* logits = (float*)carve((size_t)B_ * V_ * 4);

  auto cdiv = [](long long a, long long b) { return (int)((a + b - 1) / b); };

  // ---------- hoisted precompute ----------
  k_convT<<<cdiv((long long)A_ * D_, 256), 256, 0, stream>>>(W2, W2T, D_, A_, A_, D_);
  k_convT<<<cdiv((long long)FL_ * L_, 256), 256, 0, stream>>>(Wh, WcombT, L_, FL_, FL_, KX_);
  k_convT<<<cdiv((long long)FL_ * E_, 256), 256, 0, stream>>>(Wx + (size_t)D_ * FL_, WcombT + L_, E_, FL_, FL_, KX_);
  k_convT<<<cdiv((long long)FL_ * D_, 256), 256, 0, stream>>>(Wx, WxCT, D_, FL_, FL_, D_);
  k_convT<<<cdiv((long long)V_ * L_, 256), 256, 0, stream>>>(Wd, WdT, L_, V_, V_, L_);
  k_conv<<<cdiv((long long)B_ * S_ * D_, 256), 256, 0, stream>>>(options, Aopt, B_ * S_ * D_);

  { // opt_proj[B*S, A] = options @ W2 + b2
    int tiles = (B_ * S_ / 32) * (A_ / 16);
    k_gemm_bf16<<<cdiv(tiles, 8), 256, 0, stream>>>(Aopt, D_, W2T, b2, 1, proj, B_ * S_, A_, D_);
  }
  k_dotVa<<<cdiv(B_ * S_, 256), 256, 0, stream>>>(proj, Va, gbuf);

  float* attn_out = (float*)d_out + (size_t)B_ * T_ * V_ + 2 * (size_t)B_ * L_;
  k_attn_softmax<<<B_, S_, 0, stream>>>(gbuf, wbuf, attn_out);
  k_context<<<cdiv(B_ * D_, 256), 256, 0, stream>>>(wbuf, options, ctxbf);

  { // zc[B, 4L] = context @ Wx[:D] + bl   (loop-invariant part of z)
    int tiles = (B_ / 32) * (FL_ / 16);
    k_gemm_bf16<<<cdiv(tiles, 8), 256, 0, stream>>>(ctxbf, D_, WxCT, bl, 1, zc, B_, FL_, D_);
  }
  k_init<<<cdiv(B_ * L_, 256), 256, 0, stream>>>(h0, c0, xa, hbuf, cbuf);

  // ---------- recurrent loop ----------
  for (int t = 0; t < T_; ++t) {
    k_gather_emb<<<cdiv(B_ * E_, 256), 256, 0, stream>>>(emb, targets, t, xa);
    { // z = [h | emb_t] @ [Wh ; WxE] + zc
      int tiles = (B_ / 32) * (FL_ / 16);
      k_gemm_bf16<<<cdiv(tiles, 8), 256, 0, stream>>>(xa, KX_, WcombT, zc, 2, zbuf, B_, FL_, KX_);
    }
    k_lstm<<<cdiv(B_ * L_, 256), 256, 0, stream>>>(zbuf, cbuf, hbuf, xa);
    { // logits = h @ Wd + bd
      int tiles = (B_ / 32) * (V_ / 16);
      k_gemm_bf16<<<cdiv(tiles, 8), 256, 0, stream>>>(xa, KX_, WdT, bd, 1, logits, B_, V_, L_);
    }
    k_vsoftmax<<<B_, 256, 0, stream>>>(logits, (float*)d_out, t);
  }

  float* oh = (float*)d_out + (size_t)B_ * T_ * V_;
  float* oc = oh + (size_t)B_ * L_;
  k_final<<<cdiv(B_ * L_, 256), 256, 0, stream>>>(hbuf, cbuf, oh, oc);
}
